// HybridQNN_32719060861007
// MI455X (gfx1250) — matrix operand order repaired and checked by reading, not compiled, not run on hardware
//
#include <hip/hip_runtime.h>
#include <math.h>

typedef _Float16 v16h __attribute__((ext_vector_type(16)));
typedef float    v8f  __attribute__((ext_vector_type(8)));

// lane ^ mask within group-of-32: and=0x1F, or=0, xor=mask (offset[15]=0)
#define XOR_SWZ(v, imm) __builtin_bit_cast(float, \
    __builtin_amdgcn_ds_swizzle(__builtin_bit_cast(int, (v)), (imm)))

struct cpx { float x, y; };
__device__ __forceinline__ cpx cmul(cpx a, cpx b) {
    return { a.x * b.x - a.y * b.y, a.x * b.y + a.y * b.x };
}
__device__ __forceinline__ cpx cmadd(cpx acc, cpx a, cpx b) {
    return { acc.x + a.x * b.x - a.y * b.y, acc.y + a.x * b.y + a.y * b.x };
}

// ---------------------------------------------------------------------------
// Setup kernel (1 block, 256 threads): build U (16x16 complex) for the
// StronglyEntanglingLayers circuit, M = U^H Z0 U, real-embed to A (32x32,
// symmetric), fold in linear weight w00, and store A as f16 pre-packed in the
// WMMA B-operand lane layout for the two N-tiles:
//   bfrag[t][lane][h] = w00 * A[K][N],  K = 16*(h>=8)+8*(lane>=16)+(h&7), N = 16*t + (lane&15)
// ---------------------------------------------------------------------------
__global__ __launch_bounds__(256) void qnn_setup(const float* __restrict__ qp,
                                                 const float* __restrict__ wlin,
                                                 _Float16* __restrict__ bfrag) {
    __shared__ cpx G[8][2][2];   // Rot gates, layer-major
    __shared__ cpx U[16][16];
    __shared__ cpx M[16][16];

    const int tid = threadIdx.x;
    const int i = tid >> 4, j = tid & 15;

    if (tid < 8) {
        const int l = tid >> 2, w = tid & 3;
        const float phi = qp[(l * 4 + w) * 3 + 0];
        const float th  = qp[(l * 4 + w) * 3 + 1];
        const float om  = qp[(l * 4 + w) * 3 + 2];
        const float ct = cosf(0.5f * th), st = sinf(0.5f * th);
        const float ap = 0.5f * (phi + om), am = 0.5f * (phi - om);
        float cap, sap, cam, sam;
        __sincosf(ap, &sap, &cap);
        __sincosf(am, &sam, &cam);
        // Rot = [[ ct e^{-i ap}, -st e^{+i am} ], [ st e^{-i am}, ct e^{+i ap} ]]
        G[tid][0][0] = {  ct * cap, -ct * sap };
        G[tid][0][1] = { -st * cam, -st * sam };
        G[tid][1][0] = {  st * cam, -st * sam };
        G[tid][1][1] = {  ct * cap,  ct * sap };
    }
    U[i][j] = { (i == j) ? 1.0f : 0.0f, 0.0f };
    __syncthreads();

    // wire w maps to bit (3-w): flat index = i0*8 + i1*4 + i2*2 + i3
    for (int l = 0; l < 2; ++l) {
        for (int w = 0; w < 4; ++w) {          // Rot on wire w:  U <- (G (x) I) U
            const int g = l * 4 + w;
            const int mask = 1 << (3 - w);
            const int b  = (i & mask) ? 1 : 0;
            const int i0 = i & ~mask, i1 = i | mask;
            cpx nv = cmul(G[g][b][0], U[i0][j]);
            nv = cmadd(nv, G[g][b][1], U[i1][j]);
            __syncthreads();
            U[i][j] = nv;
            __syncthreads();
        }
        const int r = (l % 3) + 1;             // ranges = [1, 2]
        for (int w = 0; w < 4; ++w) {          // CNOT(w, (w+r)%4): row permutation
            const int cbit = 1 << (3 - w);
            const int tbit = 1 << (3 - ((w + r) & 3));
            const int src = (i & cbit) ? (i ^ tbit) : i;
            cpx nv = U[src][j];
            __syncthreads();
            U[i][j] = nv;
            __syncthreads();
        }
    }

    // M[i][j] = sum_k conj(U[k][i]) * z_k * U[k][j],  z_k = +1 if bit0(MSB)=0
    cpx m = { 0.0f, 0.0f };
    for (int k = 0; k < 16; ++k) {
        const float zk = (k & 8) ? -1.0f : 1.0f;
        const cpx a = U[k][i], b = U[k][j];
        m.x += zk * (a.x * b.x + a.y * b.y);   // Re(conj(a)*b)
        m.y += zk * (a.x * b.y - a.y * b.x);   // Im(conj(a)*b)
    }
    M[i][j] = m;
    __syncthreads();

    // A (32x32 real) = [[Mr, -Mi], [Mi, Mr]]; write f16 B-fragments, w folded in
    const float w00 = wlin[0];
    for (int e = tid; e < 1024; e += 256) {
        const int t = e >> 9;
        const int L = (e >> 4) & 31;
        const int h = e & 15;
        const int p = ((h & 8) << 1) + ((L >> 4) << 3) + (h & 7);      // K
        const int q = t * 16 + (L & 15);        // N
        float val;
        if (p < 16) val = (q < 16) ?  M[p][q].x      : -M[p][q - 16].y;
        else        val = (q < 16) ?  M[p - 16][q].y :  M[p - 16][q - 16].x;
        bfrag[e] = (_Float16)(w00 * val);
    }
}

// ---------------------------------------------------------------------------
// Main kernel: 1 wave = 16 samples. Build v_b (RX product state, 4 sincos),
// pack into WMMA A-operand layout, 2x v_wmma_f32_16x16x32_f16 -> Y = V*A,
// then z_b = <v_b, Y_b> via LDS gather + ds_swizzle xor reduction.
// ---------------------------------------------------------------------------
__global__ __launch_bounds__(256) void qnn_main(const float* __restrict__ x,
                                                const v16h* __restrict__ bfrag,
                                                const float* __restrict__ bias,
                                                float* __restrict__ out) {
    __shared__ float Vlds[8][16][33];           // per-wave V tile, padded rows

    const int tid  = threadIdx.x;
    const int wave = tid >> 5;
    const int lane = tid & 31;
    const int n    = lane & 15;                 // sample-in-tile / column id
    const int hi   = lane >> 4;                 // which lane-half
    const int tilebase = (blockIdx.x * 8 + wave) * 16;
    const int smp = tilebase + n;

    // RX embedding: s = prod_w (cos(x_w/2)|0> - i sin(x_w/2)|1>), wire0 = MSB
    const float4 xv = reinterpret_cast<const float4*>(x)[smp];
    float c[4], s[4];
    __sincosf(0.5f * xv.x, &s[0], &c[0]);
    __sincosf(0.5f * xv.y, &s[1], &c[1]);
    __sincosf(0.5f * xv.z, &s[2], &c[2]);
    __sincosf(0.5f * xv.w, &s[3], &c[3]);

    float re[16], im[16];
    re[0] = 1.0f; im[0] = 0.0f;
    int sz = 1;
#pragma unroll
    for (int w = 0; w < 4; ++w) {
#pragma unroll 16
        for (int j = 15; j >= 0; --j) {
            if (j < sz) {
                const float r0 = re[j], m0 = im[j];
                re[2 * j]     = r0 * c[w];       // bit=0: * cos
                im[2 * j]     = m0 * c[w];
                re[2 * j + 1] = m0 * s[w];       // bit=1: * (-i sin)
                im[2 * j + 1] = -r0 * s[w];
            }
        }
        sz <<= 1;
    }
    // v[k] = re[k] (k<16), im[k-16] (k>=16)

    // Stage V tile to LDS (each lane stores its half of its sample's v)
#pragma unroll
    for (int k = 0; k < 16; ++k) {
        const float sv = hi ? im[k] : re[k];
        Vlds[wave][n][16 * hi + k] = sv;
    }
    __syncthreads();

    // Pack A-operand fragment (16-bit A 16x32 layout, ISA 7.12.2):
    // lanes 0-15: h0-7 -> K=0-7,  h8-15 -> K=16-23
    // lanes16-31: h0-7 -> K=8-15, h8-15 -> K=24-31
    union { v16h v; _Float16 e[16]; } af;
#pragma unroll
    for (int h = 0; h < 8; ++h) {
        af.e[h]     = (_Float16)(hi ? re[8 + h] : re[h]);
        af.e[8 + h] = (_Float16)(hi ? im[8 + h] : im[h]);
    }

    const v16h b0 = bfrag[lane];        // A-matrix cols  0..15, B-operand layout
    const v16h b1 = bfrag[32 + lane];   // A-matrix cols 16..31
    v8f zc = { 0.f, 0.f, 0.f, 0.f, 0.f, 0.f, 0.f, 0.f };

    // Y = V * A  (two 16x16 output tiles)
    v8f d0 = __builtin_amdgcn_wmma_f32_16x16x32_f16(false, af.v, false, b0,
                                                    (short)0, zc, false, false);
    v8f d1 = __builtin_amdgcn_wmma_f32_16x16x32_f16(false, af.v, false, b1,
                                                    (short)0, zc, false, false);

    // z_M = sum_N Y[M][N] * v_M[N]; D layout: VGPR r holds M=r+8*hi, N=lane&15
    float acc[8];
#pragma unroll
    for (int r = 0; r < 8; ++r) {
        const int M = r + 8 * hi;
        acc[r] = d0[r] * Vlds[wave][M][n] + d1[r] * Vlds[wave][M][16 + n];
    }
    // reduce across the 16 lanes of each half (xor masks 1,2,4,8)
#pragma unroll
    for (int r = 0; r < 8; ++r) {
        acc[r] += XOR_SWZ(acc[r], 0x041F);   // ^1
        acc[r] += XOR_SWZ(acc[r], 0x081F);   // ^2
        acc[r] += XOR_SWZ(acc[r], 0x101F);   // ^4
        acc[r] += XOR_SWZ(acc[r], 0x201F);   // ^8
    }

    float z = 0.0f;
#pragma unroll
    for (int r = 0; r < 8; ++r) z = (n == r) ? acc[r] : z;

    if (n < 8) out[tilebase + n + 8 * hi] = z + bias[0];
}

// ---------------------------------------------------------------------------
extern "C" void kernel_launch(void* const* d_in, const int* in_sizes, int n_in,
                              void* d_out, int out_size, void* d_ws, size_t ws_size,
                              hipStream_t stream) {
    const float* x   = (const float*)d_in[0];   // [B,4]
    const float* qp  = (const float*)d_in[1];   // [2,4,3]
    const float* w   = (const float*)d_in[2];   // [1,1]
    const float* b   = (const float*)d_in[3];   // [1]
    float* out = (float*)d_out;                 // [B,1]

    _Float16* frag = (_Float16*)d_ws;           // 2*32*16 halves = 2 KB

    qnn_setup<<<1, 256, 0, stream>>>(qp, w, frag);

    const int B = in_sizes[0] / 4;              // 262144
    const int blocks = B / 128;                 // 8 waves * 16 samples per block
    qnn_main<<<blocks, 256, 0, stream>>>(x, (const v16h*)d_ws, b, out);
}
